// LSTMGapFiller_57157424775603
// MI455X (gfx1250) — compile-verified, matrix operand order checked by reading, not run on hardware
//
#include <hip/hip_runtime.h>
#include <hip/hip_bf16.h>
#include <stdint.h>

// ---------------------------------------------------------------------------
// Model dims (fixed by the reference)
// ---------------------------------------------------------------------------
#define BB    1024          // batch
#define LCTX  100           // context length
#define TT    256           // target length (device scalar; hardcoded, cannot sync-read under capture)
#define VOC   5
#define EMBD  64
#define HC    256           // context hidden
#define HG    512           // generator hidden
#define NC    (4*HC)        // 1024
#define NG    (4*HG)        // 2048

typedef __attribute__((ext_vector_type(16))) __bf16 v16bf;
typedef __attribute__((ext_vector_type(8)))  float  v8f;

// ---------------------------------------------------------------------------
// Prep: pack fp32 weight W[N,K] (row-major, gates-major) into bf16 WMMA-B
// fragment layout: P[tk][tn][lane][i],  n = tn*16 + (lane&15),
//                                       k = tk*32 + (lane>>4)*16 + i
// so each lane's 16 B elements are one contiguous 32B load per fragment.
// ---------------------------------------------------------------------------
__global__ void pack_w_kernel(const float* __restrict__ W, __bf16* __restrict__ P,
                              int N, int K) {
    int idx = blockIdx.x * blockDim.x + threadIdx.x;
    if (idx >= N * K) return;
    int i    = idx & 15;
    int lane = (idx >> 4) & 31;
    int tile = idx >> 9;                 // tk*ntiles + tn
    int ntiles = N >> 4;
    int tk = tile / ntiles;
    int tn = tile % ntiles;
    int n = tn * 16 + (lane & 15);
    int k = tk * 32 + ((lane >> 4) * 16) + i;
    P[idx] = (__bf16)W[(size_t)n * K + k];
}

// table[v][j] = sum_k emb[v,k] * Wih[j,k] + b[j]   (layer-0 input projection LUT)
__global__ void table_kernel(const float* __restrict__ emb, const float* __restrict__ Wih,
                             const float* __restrict__ bias, float* __restrict__ table, int N) {
    int idx = blockIdx.x * blockDim.x + threadIdx.x;
    if (idx >= N * VOC) return;
    int v = idx / N, j = idx % N;
    float s = bias[j];
    #pragma unroll 8
    for (int k = 0; k < EMBD; ++k) s += emb[v * EMBD + k] * Wih[j * EMBD + k];
    table[idx] = s;
}

__global__ void fill_u32_kernel(unsigned* __restrict__ p, unsigned v, int n) {
    int i = blockIdx.x * blockDim.x + threadIdx.x;
    if (i < n) p[i] = v;
}

__global__ void copy_bf16_kernel(const __bf16* __restrict__ s, __bf16* __restrict__ d, int n) {
    int i = blockIdx.x * blockDim.x + threadIdx.x;
    if (i < n) d[i] = s[i];
}

// h0 = h1 = bf16(0.5*(leftfin + rightfin))
__global__ void init_gen_kernel(const __bf16* __restrict__ lf, const __bf16* __restrict__ rf,
                                __bf16* __restrict__ h0, __bf16* __restrict__ h1) {
    int i = blockIdx.x * blockDim.x + threadIdx.x;
    if (i >= BB * HG) return;
    __bf16 v = (__bf16)(0.5f * ((float)lf[i] + (float)rf[i]));
    h0[i] = v; h1[i] = v;
}

// ---------------------------------------------------------------------------
// Core WMMA step-GEMM:  Z[1024, N] (op)= A[1024, K](bf16) @ Bp(K,N packed bf16)
//   initMode 0: Z += acc
//   initMode 1: Z  = acc + bias[n]
//   initMode 2: Z  = acc + table[tok[m*tokStride]][n]   (layer-0 fused input proj)
//
// Block: 128 threads = 4 waves. Each wave owns 4 adjacent 16x16 N-tiles
// (16M x 64N) -> block tile 16M x 256N. No LDS/barriers: all 4 waves read the
// same 1KB A tile, which hits in WGP$; weights stream from L2 (fully resident).
// Inner iteration: 1 A-fragment load, 4 B-fragment loads, 1 prefetch, 4 WMMAs.
// ---------------------------------------------------------------------------
__global__ __launch_bounds__(128)
void wmma_gemm_kernel(const __bf16* __restrict__ A, int lda,
                      const __bf16* __restrict__ Bp,
                      float* __restrict__ Z, int N, int K,
                      int initMode, const float* __restrict__ tb,
                      const int* __restrict__ tok, int tokStride) {
    const int tid    = threadIdx.x;
    const int lane   = tid & 31;
    const int wave   = tid >> 5;
    const int m0     = blockIdx.y * 16;
    const int ntiles = N >> 4;
    const int ntile0 = (blockIdx.x * 4 + wave) * 4;   // first of 4 N-tiles

    // Per-lane A fragment source: row m0+(lane&15), K-halfchunk (lane>>4)*16.
    const __bf16* aptr = A + (size_t)(m0 + (lane & 15)) * lda + ((lane >> 4) * 16);
    // Per-lane packed-B fragment source for tile (kc=0, ntile0): contiguous 32B.
    const __bf16* bbase = Bp + ((size_t)ntile0 * 32 + lane) * 16;
    const size_t  bstep = (size_t)ntiles * 32 * 16;   // one K-chunk of tiles
    const size_t  btile = (size_t)32 * 16;            // one N-tile within chunk

    v8f acc[4] = {v8f{}, v8f{}, v8f{}, v8f{}};

    for (int k0 = 0; k0 < K; k0 += 32) {
        v16bf a  = *(const v16bf*)aptr;
        v16bf b0 = *(const v16bf*)(bbase + 0 * btile);
        v16bf b1 = *(const v16bf*)(bbase + 1 * btile);
        v16bf b2 = *(const v16bf*)(bbase + 2 * btile);
        v16bf b3 = *(const v16bf*)(bbase + 3 * btile);
        // warm next K-chunk of B into cache (speculative: OOB is dropped)
        __builtin_prefetch(bbase + bstep, 0, 1);

        acc[0] = __builtin_amdgcn_wmma_f32_16x16x32_bf16(false, a, false, b0,
                                                         (short)0, acc[0], false, false);
        acc[1] = __builtin_amdgcn_wmma_f32_16x16x32_bf16(false, a, false, b1,
                                                         (short)0, acc[1], false, false);
        acc[2] = __builtin_amdgcn_wmma_f32_16x16x32_bf16(false, a, false, b2,
                                                         (short)0, acc[2], false, false);
        acc[3] = __builtin_amdgcn_wmma_f32_16x16x32_bf16(false, a, false, b3,
                                                         (short)0, acc[3], false, false);
        aptr  += 32;
        bbase += bstep;
    }

    // Epilogue. C/D layout: lanes 0-15 hold M=v, lanes 16-31 hold M=8+v; N=lane&15.
    const int nn    = lane & 15;
    const int mbase = m0 + ((lane >> 4) * 8);
    #pragma unroll
    for (int j = 0; j < 4; ++j) {
        const int n = (ntile0 + j) * 16 + nn;
        #pragma unroll
        for (int v = 0; v < 8; ++v) {
            int m = mbase + v;
            size_t idx = (size_t)m * N + n;
            float z = acc[j][v];
            if (initMode == 0)       z += Z[idx];
            else if (initMode == 1)  z += tb[n];
            else {
                int t = tok[(size_t)m * tokStride];
                z += tb[(size_t)t * N + n];
            }
            Z[idx] = z;
        }
    }
}

// ---------------------------------------------------------------------------
// LSTM gates (PyTorch order i,f,g,o):
//   c' = sig(f)*c + sig(i)*tanh(g);  h' = sig(o)*tanh(c')
// Writes h' (bf16) to recurrent buffer and optionally into a y_cat slot.
// ---------------------------------------------------------------------------
__global__ void gate_kernel(const float* __restrict__ Z, float* __restrict__ C,
                            __bf16* __restrict__ Hout,
                            __bf16* __restrict__ Ycat, int ycatStride, int Hh) {
    int idx = blockIdx.x * blockDim.x + threadIdx.x;
    if (idx >= BB * Hh) return;
    int b = idx / Hh, j = idx % Hh;
    const float* z = Z + (size_t)b * 4 * Hh;
    float zi = z[j], zf = z[Hh + j], zg = z[2 * Hh + j], zo = z[3 * Hh + j];
    float ig = 1.f / (1.f + __expf(-zi));
    float fg = 1.f / (1.f + __expf(-zf));
    float gg = tanhf(zg);
    float og = 1.f / (1.f + __expf(-zo));
    float c  = fg * C[idx] + ig * gg;
    C[idx]   = c;
    float h  = og * tanhf(c);
    __bf16 hb = (__bf16)h;
    Hout[(size_t)b * Hh + j] = hb;
    if (Ycat) Ycat[(size_t)b * ycatStride + j] = hb;
}

// logits[b,v] = h1[b]·outW[v] + outb[v]; write fp32 logits; tok[b] = argmax
__global__ void logits_kernel(const __bf16* __restrict__ H1, const float* __restrict__ W,
                              const float* __restrict__ bias, float* __restrict__ out,
                              int outStride, int* __restrict__ tok) {
    int b = blockIdx.x * blockDim.x + threadIdx.x;
    if (b >= BB) return;
    const __bf16* h = H1 + (size_t)b * HG;
    float best = -3.0e38f;
    int   bi   = 0;
    #pragma unroll
    for (int v = 0; v < VOC; ++v) {
        float s = bias[v];
        const float* w = W + v * HG;
        for (int k = 0; k < HG; ++k) s += (float)h[k] * w[k];
        out[(size_t)b * outStride + v] = s;
        if (s > best) { best = s; bi = v; }
    }
    tok[b] = bi;
}

// ---------------------------------------------------------------------------
// Host orchestration
// ---------------------------------------------------------------------------
extern "C" void kernel_launch(void* const* d_in, const int* in_sizes, int n_in,
                              void* d_out, int out_size, void* d_ws, size_t ws_size,
                              hipStream_t stream) {
    (void)in_sizes; (void)n_in; (void)out_size; (void)ws_size;

    const int*   left  = (const int*)d_in[0];
    const int*   right = (const int*)d_in[1];
    const float* emb   = (const float*)d_in[3];
    const float* c0Wih = (const float*)d_in[4];
    const float* c0Whh = (const float*)d_in[5];
    const float* c0b   = (const float*)d_in[6];
    const float* c1Wih = (const float*)d_in[7];
    const float* c1Whh = (const float*)d_in[8];
    const float* c1b   = (const float*)d_in[9];
    const float* g0Wih = (const float*)d_in[10];
    const float* g0Whh = (const float*)d_in[11];
    const float* g0b   = (const float*)d_in[12];
    const float* g1Wih = (const float*)d_in[13];
    const float* g1Whh = (const float*)d_in[14];
    const float* g1b   = (const float*)d_in[15];
    const float* outW  = (const float*)d_in[16];
    const float* outb  = (const float*)d_in[17];
    float* out = (float*)d_out;
    char*  ws  = (char*)d_ws;

    // ---- workspace layout (deterministic, recomputed each call) ----
    size_t off = 0;
    auto alloc = [&](size_t bytes) { size_t o = off; off += (bytes + 255) & ~(size_t)255; return o; };
    size_t o_pc0hh = alloc((size_t)2 * HC * NC * 2);        // ctx0 Whh packed bf16 (per dir)
    size_t o_pc1ih = alloc((size_t)2 * (2*HC) * NC * 2);    // ctx1 Wih packed (K=512)
    size_t o_pc1hh = alloc((size_t)2 * HC * NC * 2);        // ctx1 Whh packed
    size_t o_pg0hh = alloc((size_t)HG * NG * 2);            // gen0 Whh packed
    size_t o_pg1ih = alloc((size_t)HG * NG * 2);            // gen1 Wih packed
    size_t o_pg1hh = alloc((size_t)HG * NG * 2);            // gen1 Whh packed
    size_t o_t0    = alloc((size_t)2 * VOC * NC * 4);       // ctx0 vocab table (per dir)
    size_t o_tg    = alloc((size_t)VOC * NG * 4);           // gen0 vocab table
    size_t o_y0    = alloc((size_t)LCTX * BB * (2*HC) * 2); // ctx layer0 outputs (bf16)
    size_t o_y1    = alloc((size_t)LCTX * BB * (2*HC) * 2); // ctx layer1 outputs (bf16)
    size_t o_Z     = alloc((size_t)BB * NG * 4);            // gate pre-activations (fp32)
    size_t o_hctx  = alloc((size_t)BB * HC * 2);            // ctx recurrent h (bf16)
    size_t o_cctx  = alloc((size_t)BB * HC * 4);            // ctx cell state (fp32)
    size_t o_gh0   = alloc((size_t)BB * HG * 2);
    size_t o_gh1   = alloc((size_t)BB * HG * 2);
    size_t o_gc0   = alloc((size_t)BB * HG * 4);
    size_t o_gc1   = alloc((size_t)BB * HG * 4);
    size_t o_lfin  = alloc((size_t)BB * HG * 2);
    size_t o_rfin  = alloc((size_t)BB * HG * 2);
    size_t o_tok   = alloc((size_t)BB * 4);

    __bf16* pc0hh = (__bf16*)(ws + o_pc0hh);
    __bf16* pc1ih = (__bf16*)(ws + o_pc1ih);
    __bf16* pc1hh = (__bf16*)(ws + o_pc1hh);
    __bf16* pg0hh = (__bf16*)(ws + o_pg0hh);
    __bf16* pg1ih = (__bf16*)(ws + o_pg1ih);
    __bf16* pg1hh = (__bf16*)(ws + o_pg1hh);
    float*  t0    = (float*)(ws + o_t0);
    float*  tg    = (float*)(ws + o_tg);
    __bf16* y0    = (__bf16*)(ws + o_y0);
    __bf16* y1    = (__bf16*)(ws + o_y1);
    float*  Zb    = (float*)(ws + o_Z);
    __bf16* hctx  = (__bf16*)(ws + o_hctx);
    float*  cctx  = (float*)(ws + o_cctx);
    __bf16* gh0   = (__bf16*)(ws + o_gh0);
    __bf16* gh1   = (__bf16*)(ws + o_gh1);
    float*  gc0   = (float*)(ws + o_gc0);
    float*  gc1   = (float*)(ws + o_gc1);
    __bf16* lfin  = (__bf16*)(ws + o_lfin);
    __bf16* rfin  = (__bf16*)(ws + o_rfin);
    int*    tokb  = (int*)(ws + o_tok);

    auto pack = [&](const float* W, __bf16* P, int N, int K) {
        pack_w_kernel<<<dim3((N * K) / 256), dim3(256), 0, stream>>>(W, P, N, K);
    };
    auto gemm = [&](const __bf16* A, int lda, const __bf16* Bp, int N, int K,
                    int mode, const float* tb, const int* tk, int tks) {
        wmma_gemm_kernel<<<dim3(N / 256, BB / 16), dim3(128), 0, stream>>>(
            A, lda, Bp, Zb, N, K, mode, tb, tk, tks);
    };
    auto gates = [&](float* C, __bf16* Hout, __bf16* Ycat, int ystride, int Hh) {
        gate_kernel<<<dim3((BB * Hh) / 256), dim3(256), 0, stream>>>(
            Zb, C, Hout, Ycat, ystride, Hh);
    };
    auto zero = [&](void* p, size_t words) {
        fill_u32_kernel<<<dim3((unsigned)((words + 255) / 256)), dim3(256), 0, stream>>>(
            (unsigned*)p, 0u, (int)words);
    };

    // ---- prep: pack weights (bf16, B-fragment layout) + vocab tables ----
    for (int d = 0; d < 2; ++d) {
        pack(c0Whh + (size_t)d * NC * HC,      pc0hh + (size_t)d * HC * NC,       NC, HC);
        pack(c1Wih + (size_t)d * NC * (2*HC),  pc1ih + (size_t)d * (2*HC) * NC,   NC, 2 * HC);
        pack(c1Whh + (size_t)d * NC * HC,      pc1hh + (size_t)d * HC * NC,       NC, HC);
        table_kernel<<<dim3((NC * VOC + 255) / 256), dim3(256), 0, stream>>>(
            emb, c0Wih + (size_t)d * NC * EMBD, c0b + (size_t)d * NC,
            t0 + (size_t)d * VOC * NC, NC);
    }
    pack(g0Whh, pg0hh, NG, HG);
    pack(g1Wih, pg1ih, NG, HG);
    pack(g1Whh, pg1hh, NG, HG);
    table_kernel<<<dim3((NG * VOC + 255) / 256), dim3(256), 0, stream>>>(
        emb, g0Wih, g0b, tg, NG);

    // ---- bidirectional 2-layer context encoder ----
    auto run_ctx = [&](const int* toks) {
        // layer 0 (input proj is the vocab table, fused into GEMM epilogue)
        for (int d = 0; d < 2; ++d) {
            zero(hctx, (size_t)BB * HC / 2);
            zero(cctx, (size_t)BB * HC);
            for (int s = 0; s < LCTX; ++s) {
                int t = d ? (LCTX - 1 - s) : s;
                gemm(hctx, HC, pc0hh + (size_t)d * HC * NC, NC, HC,
                     /*mode=*/2, t0 + (size_t)d * VOC * NC, toks + t, LCTX);
                gates(cctx, hctx, y0 + (size_t)t * BB * (2 * HC) + d * HC, 2 * HC, HC);
            }
        }
        // layer 1 (two GEMMs per step: Wih on y_cat[t], Whh on h)
        for (int d = 0; d < 2; ++d) {
            zero(hctx, (size_t)BB * HC / 2);
            zero(cctx, (size_t)BB * HC);
            for (int s = 0; s < LCTX; ++s) {
                int t = d ? (LCTX - 1 - s) : s;
                gemm(y0 + (size_t)t * BB * (2 * HC), 2 * HC,
                     pc1ih + (size_t)d * (2 * HC) * NC, NC, 2 * HC,
                     /*mode=*/1, c1b + (size_t)d * NC, nullptr, 0);
                gemm(hctx, HC, pc1hh + (size_t)d * HC * NC, NC, HC,
                     /*mode=*/0, nullptr, nullptr, 0);
                gates(cctx, hctx, y1 + (size_t)t * BB * (2 * HC) + d * HC, 2 * HC, HC);
            }
        }
    };

    run_ctx(left);
    copy_bf16_kernel<<<dim3((BB * HG) / 256), dim3(256), 0, stream>>>(
        y1 + (size_t)(LCTX - 1) * BB * (2 * HC), lfin, BB * HG);
    run_ctx(right);
    copy_bf16_kernel<<<dim3((BB * HG) / 256), dim3(256), 0, stream>>>(
        y1, rfin, BB * HG);

    // ---- generator ----
    zero(gc0, (size_t)BB * HG);
    zero(gc1, (size_t)BB * HG);
    zero(tokb, (size_t)BB);
    init_gen_kernel<<<dim3((BB * HG) / 256), dim3(256), 0, stream>>>(lfin, rfin, gh0, gh1);

    for (int t = 0; t < TT; ++t) {
        // layer 0: z = table[tok] + h0 @ Whh^T   (single fused GEMM)
        gemm(gh0, HG, pg0hh, NG, HG, /*mode=*/2, tg, tokb, 1);
        gates(gc0, gh0, nullptr, 0, HG);
        // layer 1: z = h0 @ Wih^T + b  then  += h1 @ Whh^T
        gemm(gh0, HG, pg1ih, NG, HG, /*mode=*/1, g1b, nullptr, 0);
        gemm(gh1, HG, pg1hh, NG, HG, /*mode=*/0, nullptr, nullptr, 0);
        gates(gc1, gh1, nullptr, 0, HG);
        // logits + argmax feedback; out is [B, T, 5] fp32
        logits_kernel<<<dim3(BB / 256), dim3(256), 0, stream>>>(
            gh1, outW, outb, out + (size_t)t * VOC, TT * VOC, tokb);
    }
}
